// EngramModule_83425444757674
// MI455X (gfx1250) — matrix operand order repaired and checked by reading, not compiled, not run on hardware
//
#include <hip/hip_runtime.h>
#include <hip/hip_bf16.h>
#include <math.h>

#define B_   8
#define T_   4096
#define BT   (B_ * T_)          // 32768
#define ED   128                // EMBED_DIM
#define EED  64                 // ENGRAM_EMBED_DIM
#define NS   4                  // N_STREAMS
#define IND  256                // INPUT_DIM
#define NW   640                // val(128) + 4*key(128)
#define EPSF 1.1920929e-07f     // float32 eps
#define INV128 (1.0f / 128.0f)
#define INV_SQRT128 0.088388347648318447f

typedef float  v8f   __attribute__((ext_vector_type(8)));
typedef __bf16 v16bf __attribute__((ext_vector_type(16)));

union FragU { uint4 u[2]; v16bf v; };

__device__ __forceinline__ unsigned short f2bf(float f) {
    unsigned u = __float_as_uint(f);
    u += 0x7FFFu + ((u >> 16) & 1u);      // round to nearest even
    return (unsigned short)(u >> 16);
}

// ---------------------------------------------------------------------------
// Kernel 1: n-gram hash + embedding gather, f32 -> bf16.  One wave per (bt,g).
// ---------------------------------------------------------------------------
__global__ void __launch_bounds__(256)
k_hash_gather(const int* __restrict__ ids,
              const int* __restrict__ mult,        // (2,2,3) int32
              const float* __restrict__ emb,       // (16384,64)
              unsigned short* __restrict__ ebf) {  // (BT,256) bf16
    int tid  = threadIdx.x;
    int w    = tid >> 5;
    int lane = tid & 31;
    int task = blockIdx.x * 8 + w;        // BT*4 tasks total
    int bt   = task >> 2;
    int g    = task & 3;                  // [n0h0, n0h1, n1h0, n1h1]
    int t    = bt & (T_ - 1);

    int i2 = ids[bt];
    int i1 = (t >= 1) ? ids[bt - 1] : 0;
    int i0 = (t >= 2) ? ids[bt - 2] : 0;

    int n = g >> 1, h = g & 1;
    unsigned hash;
    if (n == 0) {  // bigram: tokens (t-1, t), mult[0][h][0..1]
        unsigned a = (unsigned)i1 * (unsigned)mult[h * 3 + 0];
        unsigned b = (unsigned)i2 * (unsigned)mult[h * 3 + 1];
        hash = (a ^ b) & 4095u;
    } else {       // trigram: tokens (t-2, t-1, t), mult[1][h][0..2]
        unsigned a = (unsigned)i0 * (unsigned)mult[6 + h * 3 + 0];
        unsigned b = (unsigned)i1 * (unsigned)mult[6 + h * 3 + 1];
        unsigned c = (unsigned)i2 * (unsigned)mult[6 + h * 3 + 2];
        hash = (a ^ b ^ c) & 4095u;
    }
    unsigned id = hash + (unsigned)g * 4096u;   // offsets [0,4096,8192,12288]

    float2 e = ((const float2*)(emb + (size_t)id * EED))[lane];
    unsigned packed = (unsigned)f2bf(e.x) | ((unsigned)f2bf(e.y) << 16);
    ((unsigned*)(ebf + (size_t)bt * IND + g * EED))[lane] = packed;
}

// ---------------------------------------------------------------------------
// Kernel 2: pack [val_W ; key_W] as bf16 (N=640, K=256), row-major (N,K).
// ---------------------------------------------------------------------------
__global__ void k_wconv(const float* __restrict__ valW,   // (128,256)
                        const float* __restrict__ keyW,   // (4,128,256)
                        unsigned short* __restrict__ wb) {
    int nrow = blockIdx.x;          // 0..639
    int k    = threadIdx.x;         // 0..255
    float v = (nrow < ED) ? valW[nrow * IND + k]
                          : keyW[(nrow - ED) * IND + k];
    wb[nrow * IND + k] = f2bf(v);
}

// ---------------------------------------------------------------------------
// Kernel 3: WMMA GEMM (BT x 256) x (256 x 128-per-group) + gate epilogue.
// grid = (BT/64, 5).  group 0 -> v_base;  groups 1..4 -> gates[.,s].
// 8 waves: wm = w&1 (M 32), wn = w>>1 (N 32); 2x2 WMMA tiles per wave.
// ---------------------------------------------------------------------------
__global__ void __launch_bounds__(256)
k_gemm(const unsigned short* __restrict__ ebf,   // (BT,256) bf16
       const unsigned short* __restrict__ wb,    // (640,256) bf16
       const float* __restrict__ x,              // (B,T,4,128)
       const float* __restrict__ val_b,          // (128)
       const float* __restrict__ key_b,          // (4,128)
       const float* __restrict__ nq,             // (4,128)
       const float* __restrict__ nk,             // (4,128)
       float* __restrict__ vbase,                // (BT,128)
       float* __restrict__ gates) {              // (BT,4)
    __shared__ float kst[64][132];     // k_raw tile (padded stride)
    __shared__ float red[64][4][4];    // per-quarter partials

    int g    = blockIdx.y;
    int m0   = blockIdx.x * 64;
    int tid  = threadIdx.x;
    int w    = tid >> 5;
    int lane = tid & 31;
    int wm   = w & 1;
    int wn   = w >> 1;
    int r    = lane & 15;
    int half = lane >> 4;

    v8f acc[2][2];
    v8f zero = {0.f, 0.f, 0.f, 0.f, 0.f, 0.f, 0.f, 0.f};
    acc[0][0] = zero; acc[0][1] = zero; acc[1][0] = zero; acc[1][1] = zero;

    int n0 = g * 128 + wn * 32;        // global W row base for this wave

    for (int kk = 0; kk < 256; kk += 32) {
        FragU a[2], b[2];
        #pragma unroll
        for (int tm = 0; tm < 2; tm++) {
            int row = m0 + wm * 32 + tm * 16 + r;
            const uint4* p = (const uint4*)(ebf + (size_t)row * IND + kk + half * 8);
            a[tm].u[0] = p[0];          // K = kk + half*8 .. +7
            a[tm].u[1] = p[2];          // K = kk + 16 + half*8 .. +7
        }
        #pragma unroll
        for (int tn = 0; tn < 2; tn++) {
            int row = n0 + tn * 16 + r;
            const uint4* p = (const uint4*)(wb + (size_t)row * IND + kk + half * 8);
            b[tn].u[0] = p[0];
            b[tn].u[1] = p[2];
        }
        #pragma unroll
        for (int tm = 0; tm < 2; tm++)
            #pragma unroll
            for (int tn = 0; tn < 2; tn++)
                acc[tm][tn] = __builtin_amdgcn_wmma_f32_16x16x32_bf16(
                    false, a[tm].v, false, b[tn].v, (short)0, acc[tm][tn],
                    false, false);
    }

    // D layout: VGPR v -> M = v + half*8 ; N = lane%16
    if (g == 0) {
        #pragma unroll
        for (int tm = 0; tm < 2; tm++)
            #pragma unroll
            for (int tn = 0; tn < 2; tn++)
                #pragma unroll
                for (int v = 0; v < 8; v++) {
                    int m = m0 + wm * 32 + tm * 16 + v + half * 8;
                    int nn = wn * 32 + tn * 16 + r;
                    vbase[(size_t)m * ED + nn] = acc[tm][tn][v] + val_b[nn];
                }
    } else {
        int s = g - 1;
        #pragma unroll
        for (int tm = 0; tm < 2; tm++)
            #pragma unroll
            for (int tn = 0; tn < 2; tn++)
                #pragma unroll
                for (int v = 0; v < 8; v++) {
                    int ml = wm * 32 + tm * 16 + v + half * 8;
                    int nl = wn * 32 + tn * 16 + r;
                    kst[ml][nl] = acc[tm][tn][v] + key_b[s * ED + nl];
                }
    }
    __syncthreads();

    if (g > 0) {
        int s    = g - 1;
        int rowl = tid >> 2;           // 0..63
        int q    = tid & 3;            // quarter: 32 cols each
        int bt   = m0 + rowl;
        const float* xrow = x + ((size_t)bt * NS + s) * ED + q * 32;
        float s_k2 = 0.f, s_x2 = 0.f, s_dot = 0.f;
        #pragma unroll 4
        for (int i = 0; i < 32; i++) {
            int c  = q * 32 + i;
            float kv = kst[rowl][c];
            float xv = xrow[i];
            s_k2 += kv * kv;
            s_x2 += xv * xv;
            s_dot += xv * nq[s * ED + c] * kv * nk[s * ED + c];
        }
        red[rowl][q][0] = s_k2;
        red[rowl][q][1] = s_x2;
        red[rowl][q][2] = s_dot;
        __syncthreads();
        if (tid < 64) {
            float k2 = red[tid][0][0] + red[tid][1][0] + red[tid][2][0] + red[tid][3][0];
            float x2 = red[tid][0][1] + red[tid][1][1] + red[tid][2][1] + red[tid][3][1];
            float dt = red[tid][0][2] + red[tid][1][2] + red[tid][2][2] + red[tid][3][2];
            float rq = rsqrtf(x2 * INV128 + EPSF);   // q rmsnorm scale
            float rk = rsqrtf(k2 * INV128 + EPSF);   // k rmsnorm scale
            float score = dt * rq * rk * INV_SQRT128;
            float gate  = 1.0f / (1.0f + __expf(-score));
            gates[(size_t)(m0 + tid) * NS + s] = gate;
        }
    }
}

// ---------------------------------------------------------------------------
// Kernel 4: v_gated = v_base*gate -> rmsnorm(convnorm) -> causal depthwise
// conv(K=4) -> silu -> + v_gated.  grid = (T/32, NS, B); 3-row halo.
// ---------------------------------------------------------------------------
__global__ void __launch_bounds__(256)
k_conv(const float* __restrict__ vbase,   // (BT,128)
       const float* __restrict__ gates,   // (BT,4)
       const float* __restrict__ convn,   // (4,128)
       const float* __restrict__ convw,   // (512,1,4)
       float* __restrict__ out) {         // (B,T,4,128)
    __shared__ float xn[35][128];
    __shared__ float vg[35][128];

    int b    = blockIdx.z;
    int s    = blockIdx.y;
    int t0   = blockIdx.x * 32;
    int tid  = threadIdx.x;
    int w    = tid >> 5;
    int lane = tid & 31;
    int c    = lane * 4;

    float4 cn = ((const float4*)(convn + s * ED))[lane];

    for (int lr = w; lr < 35; lr += 8) {
        int t = t0 - 3 + lr;
        float4 vgr = {0.f, 0.f, 0.f, 0.f};
        float  rsq = 0.f;
        if (t >= 0) {
            int bt = b * T_ + t;
            float gate = gates[(size_t)bt * NS + s];
            float4 vb = ((const float4*)(vbase + (size_t)bt * ED))[lane];
            vgr.x = vb.x * gate; vgr.y = vb.y * gate;
            vgr.z = vb.z * gate; vgr.w = vb.w * gate;
            float ss = vgr.x*vgr.x + vgr.y*vgr.y + vgr.z*vgr.z + vgr.w*vgr.w;
            #pragma unroll
            for (int m = 16; m >= 1; m >>= 1) ss += __shfl_xor(ss, m, 32);
            rsq = rsqrtf(ss * INV128 + EPSF);
        }
        float4 xr;
        xr.x = vgr.x * rsq * cn.x;
        xr.y = vgr.y * rsq * cn.y;
        xr.z = vgr.z * rsq * cn.z;
        xr.w = vgr.w * rsq * cn.w;
        *((float4*)&vg[lr][c]) = vgr;
        *((float4*)&xn[lr][c]) = xr;
    }
    __syncthreads();

    // depthwise conv weights for the 4 channels this lane owns
    const float* wp = convw + ((size_t)(s * ED + c)) * 4;
    float cw[4][4];
    #pragma unroll
    for (int i = 0; i < 4; i++)
        #pragma unroll
        for (int j = 0; j < 4; j++) cw[i][j] = wp[i * 4 + j];

    for (int olr = 3 + w; olr < 35; olr += 8) {
        int t  = t0 + olr - 3;
        int bt = b * T_ + t;
        float yv[4] = {0.f, 0.f, 0.f, 0.f};
        #pragma unroll
        for (int j = 0; j < 4; j++) {
            float4 xv = *((const float4*)&xn[olr - 3 + j][c]);
            yv[0] += xv.x * cw[0][j];
            yv[1] += xv.y * cw[1][j];
            yv[2] += xv.z * cw[2][j];
            yv[3] += xv.w * cw[3][j];
        }
        float4 vgr = *((const float4*)&vg[olr][c]);
        float4 o;
        o.x = vgr.x + yv[0] / (1.0f + __expf(-yv[0]));   // silu = y*sigmoid(y)
        o.y = vgr.y + yv[1] / (1.0f + __expf(-yv[1]));
        o.z = vgr.z + yv[2] / (1.0f + __expf(-yv[2]));
        o.w = vgr.w + yv[3] / (1.0f + __expf(-yv[3]));
        ((float4*)(out + ((size_t)bt * NS + s) * ED))[lane] = o;
    }
}

// ---------------------------------------------------------------------------
extern "C" void kernel_launch(void* const* d_in, const int* in_sizes, int n_in,
                              void* d_out, int out_size, void* d_ws, size_t ws_size,
                              hipStream_t stream) {
    (void)in_sizes; (void)n_in; (void)out_size; (void)ws_size;
    const float* x         = (const float*)d_in[0];
    const int*   input_ids = (const int*)d_in[1];
    const int*   mult      = (const int*)d_in[2];
    const float* embedding = (const float*)d_in[3];
    const float* val_W     = (const float*)d_in[4];
    const float* val_b     = (const float*)d_in[5];
    const float* key_W     = (const float*)d_in[6];
    const float* key_b     = (const float*)d_in[7];
    const float* normq_w   = (const float*)d_in[8];
    const float* normk_w   = (const float*)d_in[9];
    const float* conv_w    = (const float*)d_in[10];
    const float* convn_w   = (const float*)d_in[11];
    float* out = (float*)d_out;

    char* ws = (char*)d_ws;
    unsigned short* ebf = (unsigned short*)(ws);                 // 16,777,216 B
    unsigned short* wb  = (unsigned short*)(ws + 16777216);      //    327,680 B
    float* vbase        = (float*)(ws + 17104896);               // 16,777,216 B
    float* gates        = (float*)(ws + 33882112);               //    524,288 B

    k_hash_gather<<<(BT * 4) / 8, 256, 0, stream>>>(input_ids, mult, embedding, ebf);
    k_wconv<<<NW, 256, 0, stream>>>(val_W, key_W, wb);
    k_gemm<<<dim3(BT / 64, 5), 256, 0, stream>>>(ebf, wb, x, val_b, key_b,
                                                 normq_w, normk_w, vbase, gates);
    k_conv<<<dim3(T_ / 32, NS, B_), 256, 0, stream>>>(vbase, gates, convn_w,
                                                      conv_w, out);
}
